// UnaryAdd_75703093559676
// MI455X (gfx1250) — compile-verified, matrix operand order checked by reading, not run on hardware
//
#include <hip/hip_runtime.h>
#include <hip/hip_bf16.h>

typedef __attribute__((ext_vector_type(16))) _Float16 v16h;
typedef __attribute__((ext_vector_type(8)))  float    v8f;

#define ROWS 4096
#define COLS 16384
#define WAVES_PER_BLOCK 16
#define THREADS (WAVES_PER_BLOCK * 32)            // 512 threads = 16 wave32 waves
#define ROWS_PER_WAVE (ROWS / WAVES_PER_BLOCK)    // 256 rows -> 8 iterations of K=32
#define COLS_PER_BLOCK 32                          // two 16-col WMMA tiles: full 128B lines/wave

// bit = ( round((p+1)*128) > sobol(rng_idx) ),  sobol(i) = bitreverse8(i ^ (i>>1))
// brev(ri ^ (ri>>1)) == t ^ (t<<1) with t = brev(ri); byte3 of that is the
// 8-bit-reversed gray code -> (float)(u >> 24) lowers to v_cvt_f32_ubyte3.
__device__ __forceinline__ _Float16 sc_bit(float p, int ri) {
    unsigned t = __brev((unsigned)ri);
    unsigned u = t ^ (t << 1);
    float thr  = (float)(u >> 24);                               // sobol threshold
    float src  = rintf(__builtin_fmaf(p, 128.0f, 128.0f));       // == round((p+1)*128), RNE
    return (src > thr) ? (_Float16)1.0f : (_Float16)0.0f;
}

__global__ __launch_bounds__(THREADS) void sc_forward_wmma(
    const float* __restrict__ prob,
    const int*   __restrict__ rng_idx,
    const float* __restrict__ accum,
    float*       __restrict__ out)
{
    const int lane = threadIdx.x & 31;
    const int wave = threadIdx.x >> 5;
    const int col0 = blockIdx.x * COLS_PER_BLOCK;
    const int n    = lane & 15;              // column within a 16-wide tile
    const int kgrp = (lane >> 4) * 16;       // lanes 0-15: K 0-15, lanes 16-31: K 16-31

    __shared__ float partial[WAVES_PER_BLOCK][COLS_PER_BLOCK];

    // A = all-ones 16x32 f16 fragment (value is layout-invariant).
    v16h a;
    #pragma unroll
    for (int i = 0; i < 16; ++i) a[i] = (_Float16)1.0f;

    v8f c0 = {};   // counts for columns col0 +  0..15
    v8f c1 = {};   // counts for columns col0 + 16..31

    const int row_begin = wave * ROWS_PER_WAVE;
    const int row_end   = row_begin + ROWS_PER_WAVE;

    #pragma unroll 1
    for (int r0 = row_begin; r0 < row_end; r0 += 32) {
        union { v16h v; _Float16 h[16]; } b0, b1;
        const size_t base = (size_t)(r0 + kgrp) * COLS + (size_t)(col0 + n);
        #pragma unroll
        for (int e = 0; e < 16; ++e) {
            const size_t off = base + (size_t)e * COLS;   // folds into 24-bit IOFFSET
            const float p0 = prob[off];
            const float p1 = prob[off + 16];              // adjacent 64B: same 128B line
            const int   i0 = rng_idx[off];
            const int   i1 = rng_idx[off + 16];
            b0.h[e] = sc_bit(p0, i0);
            b1.h[e] = sc_bit(p1, i1);
        }
        // D = ones * bits + C  => every row of D is the running column count.
        c0 = __builtin_amdgcn_wmma_f32_16x16x32_f16(false, a, false, b0.v,
                                                    (short)0, c0, false, false);
        c1 = __builtin_amdgcn_wmma_f32_16x16x32_f16(false, a, false, b1.v,
                                                    (short)0, c1, false, false);
    }

    // C/D layout: VGPR 0 of lane L holds (M, N = L%16); all M-rows equal the count,
    // so lanes 0-15 publish tile 0 and lanes 16-31 publish tile 1 — full-wave store.
    if (lane < 16) partial[wave][n]      = c0[0];
    else           partial[wave][16 + n] = c1[0];
    __syncthreads();

    if (threadIdx.x < COLS_PER_BLOCK) {
        float cnt = 0.0f;
        #pragma unroll
        for (int w = 0; w < WAVES_PER_BLOCK; ++w)
            cnt += partial[w][threadIdx.x];
        const float accv = accum[col0 + threadIdx.x] + cnt;
        out[col0 + threadIdx.x] = (accv >= (float)ROWS) ? 1.0f : 0.0f;
    }
}

extern "C" void kernel_launch(void* const* d_in, const int* in_sizes, int n_in,
                              void* d_out, int out_size, void* d_ws, size_t ws_size,
                              hipStream_t stream) {
    // setup_inputs() order: st_in_prob, accumulator, input_bits (unused), rng_idx
    const float* prob  = (const float*)d_in[0];
    const float* accum = (const float*)d_in[1];
    const int*   rng   = (const int*)d_in[3];
    float*       out   = (float*)d_out;

    dim3 grid(COLS / COLS_PER_BLOCK);   // 512 blocks of 32 columns
    dim3 block(THREADS);                // 16 wave32 waves per block
    hipLaunchKernelGGL(sc_forward_wmma, grid, block, 0, stream,
                       prob, rng, accum, out);
}
